// SimMatrixLoss_29343216566713
// MI455X (gfx1250) — matrix operand order repaired and checked by reading, not compiled, not run on hardware
//
#include <hip/hip_runtime.h>
#include <hip/hip_bf16.h>

typedef __attribute__((ext_vector_type(16))) __bf16        v16bf;
typedef __attribute__((ext_vector_type(8)))  float         v8f;
typedef __attribute__((ext_vector_type(8)))  unsigned int  v8u;

#define NS    512
#define DDIM  256
#define BATCH 128

// ---------- helpers ----------
__device__ __forceinline__ float wave_reduce(float v) {
  #pragma unroll
  for (int off = 16; off; off >>= 1) v += __shfl_xor(v, off, 32);
  return v;
}

// round-to-nearest-even f32 -> bf16 (result kept in high 16 bits)
__device__ __forceinline__ unsigned bf_rne_bits(float f) {
  unsigned u = __float_as_uint(f);
  return u + 0x7FFFu + ((u >> 16) & 1u);
}

// pack two floats -> dword of bf16 hi parts + dword of bf16 residual parts
__device__ __forceinline__ void pack2(float f0, float f1, unsigned &hi, unsigned &lo) {
  unsigned r0 = bf_rne_bits(f0), r1 = bf_rne_bits(f1);
  hi = (r0 >> 16) | (r1 & 0xFFFF0000u);
  float h0 = __uint_as_float(r0 & 0xFFFF0000u);
  float h1 = __uint_as_float(r1 & 0xFFFF0000u);
  unsigned t0 = bf_rne_bits(f0 - h0), t1 = bf_rne_bits(f1 - h1);
  lo = (t0 >> 16) | (t1 & 0xFFFF0000u);
}

// A-fragment k order used here (16x32, same as B): lanes 0-15 K=0..15, lanes 16-31 K=16..31 (consecutive)
// p already points at row + kb*32 + lh*16 (bf16 elements)
__device__ __forceinline__ v16bf frag_bf_A(const unsigned short* p) {
  uint4 a0 = *(const uint4*)(p);
  uint4 a1 = *(const uint4*)(p + 8);
  v8u v = {a0.x, a0.y, a0.z, a0.w, a1.x, a1.y, a1.z, a1.w};
  return __builtin_bit_cast(v16bf, v);
}
// B-fragment k order used here (32x16, same as A): lanes 0-15 K=0..15, lanes 16-31 K=16..31 (consecutive)
// p already points at row + kb*32 + lh*16
__device__ __forceinline__ v16bf frag_bf_B(const unsigned short* p) {
  uint4 b0 = *(const uint4*)(p);
  uint4 b1 = *(const uint4*)(p + 8);
  v8u v = {b0.x, b0.y, b0.z, b0.w, b1.x, b1.y, b1.z, b1.w};
  return __builtin_bit_cast(v16bf, v);
}
// fallback: build hi/lo fragments straight from fp32
__device__ __forceinline__ void frag_f32_A(const float* p, v16bf& h, v16bf& l) {
  float4 x0 = *(const float4*)(p + 0),  x1 = *(const float4*)(p + 4);
  float4 x2 = *(const float4*)(p + 8), x3 = *(const float4*)(p + 12);
  unsigned H[8], L[8];
  pack2(x0.x, x0.y, H[0], L[0]); pack2(x0.z, x0.w, H[1], L[1]);
  pack2(x1.x, x1.y, H[2], L[2]); pack2(x1.z, x1.w, H[3], L[3]);
  pack2(x2.x, x2.y, H[4], L[4]); pack2(x2.z, x2.w, H[5], L[5]);
  pack2(x3.x, x3.y, H[6], L[6]); pack2(x3.z, x3.w, H[7], L[7]);
  v8u vh = {H[0], H[1], H[2], H[3], H[4], H[5], H[6], H[7]};
  v8u vl = {L[0], L[1], L[2], L[3], L[4], L[5], L[6], L[7]};
  h = __builtin_bit_cast(v16bf, vh);
  l = __builtin_bit_cast(v16bf, vl);
}
__device__ __forceinline__ void frag_f32_B(const float* p, v16bf& h, v16bf& l) {
  float4 x0 = *(const float4*)(p + 0), x1 = *(const float4*)(p + 4);
  float4 x2 = *(const float4*)(p + 8), x3 = *(const float4*)(p + 12);
  unsigned H[8], L[8];
  pack2(x0.x, x0.y, H[0], L[0]); pack2(x0.z, x0.w, H[1], L[1]);
  pack2(x1.x, x1.y, H[2], L[2]); pack2(x1.z, x1.w, H[3], L[3]);
  pack2(x2.x, x2.y, H[4], L[4]); pack2(x2.z, x2.w, H[5], L[5]);
  pack2(x3.x, x3.y, H[6], L[6]); pack2(x3.z, x3.w, H[7], L[7]);
  v8u vh = {H[0], H[1], H[2], H[3], H[4], H[5], H[6], H[7]};
  v8u vl = {L[0], L[1], L[2], L[3], L[4], L[5], L[6], L[7]};
  h = __builtin_bit_cast(v16bf, vh);
  l = __builtin_bit_cast(v16bf, vl);
}

// ---------- kernel 0: zero accumulators ----------
__global__ void k_init(float* ws) {
  int t = threadIdx.x;
  if (t <= BATCH) ws[t] = 0.0f;   // partial[0..127], denomsq at [128]
}

// ---------- kernel 1: preconvert fp32 -> bf16 hi/lo arrays ----------
__global__ void k_cvt(const float* __restrict__ dv,
                      unsigned short* __restrict__ hi,
                      unsigned short* __restrict__ lo, int n) {
  int i = blockIdx.x * blockDim.x + threadIdx.x;
  if (i >= n) return;
  float f = dv[i];
  unsigned r = bf_rne_bits(f);
  unsigned short h = (unsigned short)(r >> 16);
  float hf = __uint_as_float((unsigned)h << 16);
  unsigned rl = bf_rne_bits(f - hf);
  hi[i] = h;
  lo[i] = (unsigned short)(rl >> 16);
}

// ---------- kernel 2: row squared-norms (wave per row, fp32 exact) ----------
__global__ void k_n2(const float* __restrict__ dv, float* __restrict__ n2) {
  int wid  = (blockIdx.x * blockDim.x + threadIdx.x) >> 5;   // row in [0, B*Ns)
  int lane = threadIdx.x & 31;
  const float* p = dv + (size_t)wid * DDIM;
  float acc = 0.f;
  #pragma unroll
  for (int t = 0; t < 8; ++t) { float v = p[lane + 32 * t]; acc += v * v; }
  acc = wave_reduce(acc);
  if (lane == 0) n2[wid] = acc;
}

// ---------- kernel 3: denom^2 = ||W - I||_F^2 ----------
__global__ void k_denom(const float* __restrict__ S, float* __restrict__ denomsq) {
  int idx = blockIdx.x * blockDim.x + threadIdx.x;  // 65536 threads, 4 elems each
  float acc = 0.f;
  #pragma unroll
  for (int t = 0; t < 4; ++t) {
    int e = idx + t * 65536;
    int i = e >> 9, j = e & (NS - 1);
    float s = S[e];
    float w = (s > 0.f) ? 1.f : 0.f;
    float d = w - ((i == j) ? 1.f : 0.f);
    acc += d * d;
  }
  acc = wave_reduce(acc);
  if ((threadIdx.x & 31) == 0) atomicAdd(denomsq, acc);
}

// ---------- main kernel ----------
// Block = 256 threads (8 waves) = one 16-row i-tile x 256 columns.
// Each wave owns TWO j-tiles (two independent WMMA accumulation chains,
// A hi/lo fragments reused across both -> 12 b128 loads per 6 WMMAs).
template <bool PRE>
__global__ void k_main(const float* __restrict__ dv,
                       const unsigned short* __restrict__ hi,
                       const unsigned short* __restrict__ lo,
                       const float* __restrict__ S,
                       const float* __restrict__ n2,
                       float* __restrict__ partial) {
  const int lane = threadIdx.x & 31;
  const int wave = threadIdx.x >> 5;
  const int bid  = blockIdx.x;
  const int b    = bid >> 6;          // / 64
  const int rem  = bid & 63;
  const int it   = rem >> 1;          // 0..31  i-tile
  const int jg   = rem & 1;           // 0..1   j-group (256 cols each)
  const int i0   = it << 4;
  const int jt0  = (jg << 4) + wave;      // first j-tile
  const int jt1  = (jg << 4) + 8 + wave;  // second j-tile
  const int j0a  = jt0 << 4;
  const int j0b  = jt1 << 4;
  const int lrow = lane & 15;
  const int lh   = lane >> 4;

  const size_t rowA  = (size_t)(b * NS + i0  + lrow) * DDIM;
  const size_t rowBa = (size_t)(b * NS + j0a + lrow) * DDIM;
  const size_t rowBb = (size_t)(b * NS + j0b + lrow) * DDIM;

  v8f acc0 = {}, acc1 = {};
  #pragma unroll 2
  for (int kb = 0; kb < 8; ++kb) {
    v16bf ah, al, b0h, b0l, b1h, b1l;
    if constexpr (PRE) {
      const size_t ka  = rowA  + kb * 32 + lh * 16;
      const size_t kba = rowBa + kb * 32 + lh * 16;
      const size_t kbb = rowBb + kb * 32 + lh * 16;
      ah  = frag_bf_A(hi + ka);   al  = frag_bf_A(lo + ka);
      b0h = frag_bf_B(hi + kba);  b0l = frag_bf_B(lo + kba);
      b1h = frag_bf_B(hi + kbb);  b1l = frag_bf_B(lo + kbb);
    } else {
      frag_f32_A(dv + rowA  + kb * 32 + lh * 16, ah,  al);
      frag_f32_B(dv + rowBa + kb * 32 + lh * 16, b0h, b0l);
      frag_f32_B(dv + rowBb + kb * 32 + lh * 16, b1h, b1l);
    }
    // G ~= a_lo*b_hi + a_hi*b_lo + a_hi*b_hi  (near-fp32); two independent chains
    acc0 = __builtin_amdgcn_wmma_f32_16x16x32_bf16(false, al, false, b0h, (short)0, acc0, false, false);
    acc1 = __builtin_amdgcn_wmma_f32_16x16x32_bf16(false, al, false, b1h, (short)0, acc1, false, false);
    acc0 = __builtin_amdgcn_wmma_f32_16x16x32_bf16(false, ah, false, b0l, (short)0, acc0, false, false);
    acc1 = __builtin_amdgcn_wmma_f32_16x16x32_bf16(false, ah, false, b1l, (short)0, acc1, false, false);
    acc0 = __builtin_amdgcn_wmma_f32_16x16x32_bf16(false, ah, false, b0h, (short)0, acc0, false, false);
    acc1 = __builtin_amdgcn_wmma_f32_16x16x32_bf16(false, ah, false, b1h, (short)0, acc1, false, false);
  }

  // Epilogue. C/D layout: VGPR r -> M = r + 8*lh, N = lane & 15.
  float sum = 0.f;
  #pragma unroll
  for (int tile = 0; tile < 2; ++tile) {
    const int   j   = (tile ? j0b : j0a) + lrow;
    const v8f&  acc = tile ? acc1 : acc0;
    const float n2j = n2[b * NS + j];
    #pragma unroll
    for (int r = 0; r < 8; ++r) {
      const int i   = i0 + r + lh * 8;
      const float g = acc[r];
      float sq = n2[b * NS + i] + n2j - 2.f * g;
      sq = fmaxf(sq, 0.f);
      const bool diag = (i == j);
      const float dist = (sq > 0.f) ? __builtin_sqrtf(sq) : 0.f;
      const float kd   = __expf(-dist);          // GAMMA = 1
      const float kdt  = diag ? 0.f : kd;        // Kd * (1 - eye)
      const float s    = S[i * NS + j];
      const float st   = s - (diag ? 1.f : 0.f); // S - MAX_SIMILARITY*eye
      const float w    = (s > 0.f) ? 1.f : 0.f;
      const float d    = w * (kdt - st);
      sum += d * d;
    }
  }
  sum = wave_reduce(sum);
  if (lane == 0) atomicAdd(&partial[b], sum);
}

// ---------- finalize: sum_b 2*sqrt(partial_b)/sqrt(denomsq) ----------
__global__ void k_final(const float* __restrict__ partial,
                        const float* __restrict__ denomsq,
                        float* __restrict__ out) {
  __shared__ float sm[4];
  int t = threadIdx.x;               // 128 threads
  float v = __builtin_sqrtf(fmaxf(partial[t], 0.f));
  float w = wave_reduce(v);
  if ((t & 31) == 0) sm[t >> 5] = w;
  __syncthreads();
  if (t == 0) {
    float tot = sm[0] + sm[1] + sm[2] + sm[3];
    out[0] = 2.f * tot / __builtin_sqrtf(denomsq[0]);
  }
}

extern "C" void kernel_launch(void* const* d_in, const int* in_sizes, int n_in,
                              void* d_out, int out_size, void* d_ws, size_t ws_size,
                              hipStream_t stream) {
  const float* dv = (const float*)d_in[0];   // [128, 512, 256] f32
  const float* S  = (const float*)d_in[1];   // [512, 512] f32
  float* out      = (float*)d_out;           // scalar f32
  float* ws       = (float*)d_ws;

  float* partial  = ws;                      // 128 floats
  float* denomsq  = ws + 128;                // 1 float
  float* n2       = ws + 256;                // 65536 floats (256 KB)
  unsigned short* hi = (unsigned short*)((char*)d_ws + (size_t)(1u << 20));            // 32 MB
  unsigned short* lo = (unsigned short*)((char*)d_ws + (size_t)(1u << 20) + (size_t)(32u << 20));
  const int nelem = BATCH * NS * DDIM;       // 16,777,216
  const bool pre = ws_size >= ((size_t)(1u << 20) + (size_t)(64u << 20) + 4096);

  k_init<<<1, 256, 0, stream>>>(ws);
  k_n2<<<(BATCH * NS) / 8, 256, 0, stream>>>(dv, n2);            // 8192 blocks, wave/row
  k_denom<<<256, 256, 0, stream>>>(S, denomsq);
  if (pre) {
    k_cvt<<<nelem / 256, 256, 0, stream>>>(dv, hi, lo, nelem);
    k_main<true><<<BATCH * 64, 256, 0, stream>>>(dv, hi, lo, S, n2, partial);
  } else {
    k_main<false><<<BATCH * 64, 256, 0, stream>>>(dv, nullptr, nullptr, S, n2, partial);
  }
  k_final<<<1, 128, 0, stream>>>(partial, denomsq, out);
}
